// Layer_2851858284854
// MI455X (gfx1250) — matrix operand order repaired and checked by reading, not compiled, not run on hardware
//
#include <hip/hip_runtime.h>
#include <hip/hip_bf16.h>

// ---------------------------------------------------------------------------
// Model dimensions (from reference)
// ---------------------------------------------------------------------------
#define B_   4
#define L_   1024
#define DM_  2048
#define H_   32
#define KV_  4
#define HD_  128
#define G_   (H_ / KV_)   // 8
#define E_   16
#define TOPK_ 8
#define MI_  128
#define N_   (B_ * L_)    // 4096 tokens

typedef __bf16 bf16_t;
typedef __attribute__((ext_vector_type(16))) __bf16 v16bf;
typedef __attribute__((ext_vector_type(8)))  float  v8f;

union ABFrag { v16bf v; unsigned int u[8]; };

__device__ __forceinline__ bf16_t f2bf(float f) {
    unsigned int u = __builtin_bit_cast(unsigned int, f);
    unsigned int r = (u + 0x7FFFu + ((u >> 16) & 1u)) >> 16;
    return __builtin_bit_cast(bf16_t, (unsigned short)r);
}
__device__ __forceinline__ float bf2f(bf16_t b) {
    unsigned int u = ((unsigned int)__builtin_bit_cast(unsigned short, b)) << 16;
    return __builtin_bit_cast(float, u);
}
__device__ __forceinline__ v8f v8f_zero() {
    v8f z = {0.f, 0.f, 0.f, 0.f, 0.f, 0.f, 0.f, 0.f};
    return z;
}

// CDNA5 async global->LDS copy of 16 bytes (tracked by ASYNCcnt).
// lds_off: LDS byte offset (low 32 bits of a flat pointer into LDS);
// gaddr:   64-bit global address. EXEC must be all ones at the call sites.
__device__ __forceinline__ void async_copy_b128(unsigned lds_off, const void* gaddr) {
    asm volatile("global_load_async_to_lds_b128 %0, %1, off"
                 :: "v"(lds_off), "v"((unsigned long long)(size_t)gaddr)
                 : "memory");
}
// Drain all outstanding async copies.
__device__ __forceinline__ void wait_asynccnt_0() {
    asm volatile("s_wait_asynccnt 0" ::: "memory");
}
// Leave the newest 4 async copies (next stage) in flight; async loads complete
// in order, so ASYNCcnt <= 4 means the previous stage's 4 ops have landed.
__device__ __forceinline__ void wait_asynccnt_4() {
    asm volatile("s_wait_asynccnt 4" ::: "memory");
}

// ---------------------------------------------------------------------------
// fp32 -> bf16 conversion (weights / activations)
// ---------------------------------------------------------------------------
__global__ void k_f32_to_bf16(const float* __restrict__ in, bf16_t* __restrict__ out, int n) {
    int i = blockIdx.x * blockDim.x + threadIdx.x;
    if (i < n) out[i] = f2bf(in[i]);
}

// ---------------------------------------------------------------------------
// Row-wise RMSNorm: out_bf16[row] = x[row] * rsqrt(mean(x^2)+eps) * w
// ---------------------------------------------------------------------------
__global__ void k_rmsnorm_bf16(const float* __restrict__ x, const float* __restrict__ w,
                               bf16_t* __restrict__ out) {
    const int row = blockIdx.x;
    const float* xr = x + (size_t)row * DM_;
    float ss = 0.f;
    for (int i = threadIdx.x; i < DM_; i += blockDim.x) { float v = xr[i]; ss += v * v; }
    __shared__ float red[8];
    #pragma unroll
    for (int o = 16; o > 0; o >>= 1) ss += __shfl_xor(ss, o);
    const int wid = threadIdx.x >> 5, lane = threadIdx.x & 31;
    if (lane == 0) red[wid] = ss;
    __syncthreads();
    if (wid == 0) {
        float v = (lane < 8) ? red[lane] : 0.f;
        #pragma unroll
        for (int o = 4; o > 0; o >>= 1) v += __shfl_xor(v, o);
        if (lane == 0) red[0] = v;
    }
    __syncthreads();
    const float inv = rsqrtf(red[0] / (float)DM_ + 1e-6f);
    for (int i = threadIdx.x; i < DM_; i += blockDim.x)
        out[(size_t)row * DM_ + i] = f2bf(xr[i] * inv * w[i]);
}

// ---------------------------------------------------------------------------
// Simple WMMA GEMM (one 16x16 tile per wave) — used for skinny M (MoE gate/up,
// M=128, weights L2-resident, maximal grid parallelism).
// C[N,M] = A[N,K](bf16) * W[M,K](bf16)^T
// ---------------------------------------------------------------------------
__global__ void __launch_bounds__(256)
k_wmma_gemm_nt(const bf16_t* __restrict__ A, const bf16_t* __restrict__ W,
               float* __restrict__ C, int Nrows, int Mcols, int K,
               const float* __restrict__ residual,
               const float* __restrict__ rowscale, int rs_stride, int accumulate) {
    const int lane  = threadIdx.x & 31;
    const int wid   = threadIdx.x >> 5;
    const int tiles_m = Mcols >> 4;
    const int tiles_n = Nrows >> 4;
    const int tile = blockIdx.x * 8 + wid;
    if (tile >= tiles_m * tiles_n) return;
    const int tm = tile % tiles_m;
    const int tn = tile / tiles_m;
    const int r = lane & 15, khalf = lane >> 4;

    const bf16_t* arow = A + (size_t)(tn * 16 + r) * K;
    const bf16_t* brow = W + (size_t)(tm * 16 + r) * K;

    v8f acc = v8f_zero();
    for (int k0 = 0; k0 < K; k0 += 32) {
        ABFrag a, b;
        #pragma unroll
        for (int j = 0; j < 8; ++j) {
            const int ka = k0 + ((j < 4) ? (2 * j) : (16 + 2 * (j - 4))) + 8 * khalf;
            a.u[j] = *(const unsigned int*)(arow + ka);
            const int kb = k0 + ((j < 4) ? (2 * j) : (16 + 2 * (j - 4))) + 8 * khalf;
            b.u[j] = *(const unsigned int*)(brow + kb);
        }
        __builtin_prefetch(arow + k0 + 64, 0, 1);
        __builtin_prefetch(brow + k0 + 64, 0, 1);
        acc = __builtin_amdgcn_wmma_f32_16x16x32_bf16(false, a.v, false, b.v,
                                                      (short)0, acc, false, false);
    }
    #pragma unroll
    for (int j = 0; j < 8; ++j) {
        const int gn = tn * 16 + j + 8 * khalf;
        const int gm = tm * 16 + r;
        const size_t idx = (size_t)gn * Mcols + gm;
        float v = acc[j];
        if (rowscale) v *= rowscale[(size_t)gn * rs_stride];
        if (accumulate) C[idx] += v;
        else            C[idx] = v + (residual ? residual[idx] : 0.f);
    }
}

// ---------------------------------------------------------------------------
// Blocked WMMA GEMM: 128x128 C block per 256-thread workgroup.
// 8 waves in a 2(N) x 4(M) grid; each wave computes 64x32 = eight 16x16 tiles.
// A(128x32)/B(128x32) bf16 tiles double-buffered in LDS, filled through CDNA5
// async global->LDS b128 copies. Stage k+1 is issued before computing stage k;
// s_wait_asynccnt 4 drains only the previous stage (async loads complete in
// order), overlapping copy latency with WMMA. ~64 FLOP/B from global.
// Requires Nrows % 128 == 0 && Mcols % 128 == 0 && K % 32 == 0.
// ---------------------------------------------------------------------------
#define LDA_PAD 40  // LDS row stride in elements: 32 + 8 pad = 80 bytes (16B aligned)

__global__ void __launch_bounds__(256)
k_wmma_gemm_blk(const bf16_t* __restrict__ A, const bf16_t* __restrict__ W,
                float* __restrict__ C, int Nrows, int Mcols, int K,
                const float* __restrict__ residual,
                const float* __restrict__ rowscale, int rs_stride, int accumulate) {
    const int lane = threadIdx.x & 31;
    const int wid  = threadIdx.x >> 5;
    const int wn   = wid >> 2;          // 0..1  (N direction)
    const int wm   = wid & 3;           // 0..3  (M direction)
    const int bn   = blockIdx.y * 128;
    const int bm   = blockIdx.x * 128;
    const int r = lane & 15, khalf = lane >> 4;

    __shared__ __align__(16) bf16_t As[2][128 * LDA_PAD];
    __shared__ __align__(16) bf16_t Bs[2][128 * LDA_PAD];

    v8f acc[4][2];
    #pragma unroll
    for (int i = 0; i < 4; ++i)
        #pragma unroll
        for (int m = 0; m < 2; ++m) acc[i][m] = v8f_zero();

    const int t = threadIdx.x;
    // Issue one stage of async copies (4 per thread: 2 segments x {A,B}).
    auto stage = [&](int buf, int kk) {
        #pragma unroll
        for (int i = 0; i < 2; ++i) {
            const int seg = t + i * 256;        // 0..511 : 16B segments of a 128x32 tile
            const int row = seg >> 2;           // 0..127
            const int col = (seg & 3) * 8;      // element column
            const bf16_t* ga = A + (size_t)(bn + row) * K + kk + col;
            const bf16_t* gb = W + (size_t)(bm + row) * K + kk + col;
            async_copy_b128((unsigned)(size_t)&As[buf][row * LDA_PAD + col], ga);
            async_copy_b128((unsigned)(size_t)&Bs[buf][row * LDA_PAD + col], gb);
        }
    };

    stage(0, 0);
    int cur = 0;
    for (int k0 = 0; k0 < K; k0 += 32) {
        const bool more = (k0 + 32) < K;
        if (more) stage(cur ^ 1, k0 + 32);   // prefetch next stage into other buffer
        if (more) wait_asynccnt_4();         // previous stage landed; next 4 in flight
        else      wait_asynccnt_0();
        __syncthreads();

        ABFrag af[4], bfr[2];
        #pragma unroll
        for (int i = 0; i < 4; ++i) {
            const int arow = wn * 64 + i * 16 + r;
            #pragma unroll
            for (int j = 0; j < 8; ++j) {
                const int ka = ((j < 4) ? (2 * j) : (16 + 2 * (j - 4))) + 8 * khalf;
                af[i].u[j] = *(const unsigned int*)&As[cur][arow * LDA_PAD + ka];
            }
        }
        #pragma unroll
        for (int m = 0; m < 2; ++m) {
            const int brow = wm * 32 + m * 16 + r;
            #pragma unroll
            for (int j = 0; j < 8; ++j) {
                const int kb = ((j < 4) ? (2 * j) : (16 + 2 * (j - 4))) + 8 * khalf;
                bfr[m].u[j] = *(const unsigned int*)&Bs[cur][brow * LDA_PAD + kb];
            }
        }
        #pragma unroll
        for (int i = 0; i < 4; ++i)
            #pragma unroll
            for (int m = 0; m < 2; ++m)
                acc[i][m] = __builtin_amdgcn_wmma_f32_16x16x32_bf16(
                    false, af[i].v, false, bfr[m].v, (short)0, acc[i][m], false, false);
        __syncthreads();   // all reads of buffer `cur` done before it is refilled
        cur ^= 1;
    }

    #pragma unroll
    for (int i = 0; i < 4; ++i)
        #pragma unroll
        for (int m = 0; m < 2; ++m)
            #pragma unroll
            for (int j = 0; j < 8; ++j) {
                const int gn = bn + wn * 64 + i * 16 + j + 8 * khalf;
                const int gm = bm + wm * 32 + m * 16 + r;
                const size_t idx = (size_t)gn * Mcols + gm;
                float v = acc[i][m][j];
                if (rowscale) v *= rowscale[(size_t)gn * rs_stride];
                if (accumulate) C[idx] += v;
                else            C[idx] = v + (residual ? residual[idx] : 0.f);
            }
}

// ---------------------------------------------------------------------------
// Per-head RMSNorm + RoPE; relayout [N, nh*HD] f32 -> [B, nh, L, HD] bf16.
// ---------------------------------------------------------------------------
__global__ void k_qk_norm_rope(const float* __restrict__ src, const float* __restrict__ nw,
                               const float* __restrict__ cosb, const float* __restrict__ sinb,
                               bf16_t* __restrict__ dst, int nh) {
    const int n = blockIdx.x, h = blockIdx.y, d = threadIdx.x;
    const int b = n / L_, l = n % L_;
    const float v = src[(size_t)n * nh * HD_ + h * HD_ + d];
    float ss = v * v;
    #pragma unroll
    for (int o = 16; o > 0; o >>= 1) ss += __shfl_xor(ss, o);
    __shared__ float red[4];
    if ((threadIdx.x & 31) == 0) red[threadIdx.x >> 5] = ss;
    __syncthreads();
    const float tot = red[0] + red[1] + red[2] + red[3];
    const float xn = v * rsqrtf(tot / (float)HD_ + 1e-6f) * nw[d];
    __shared__ float sh[HD_];
    sh[d] = xn;
    __syncthreads();
    const float other = (d < HD_ / 2) ? -sh[d + HD_ / 2] : sh[d - HD_ / 2];
    const size_t ci = (size_t)(b * L_ + l) * HD_ + d;
    const float out = cosb[ci] * xn + sinb[ci] * other;
    dst[(((size_t)b * nh + h) * L_ + l) * HD_ + d] = f2bf(out);
}

// V: [N, KV*HD] f32 -> transposed [B, KV, HD, L] bf16 (so P*V B-frags load as dwords)
__global__ void k_v_transpose(const float* __restrict__ vf, bf16_t* __restrict__ vt) {
    const int n = blockIdx.x, h = blockIdx.y, d = threadIdx.x;
    const int b = n / L_, l = n % L_;
    vt[(((size_t)b * KV_ + h) * HD_ + d) * L_ + l] =
        f2bf(vf[(size_t)n * KV_ * HD_ + h * HD_ + d]);
}

// ---------------------------------------------------------------------------
// Flash attention (causal, GQA). 4 waves / block, each wave = 16 query rows.
// QK^T and P*V on v_wmma_f32_16x16x32_bf16; online softmax via 16-lane shuffles.
// ---------------------------------------------------------------------------
__global__ void __launch_bounds__(128)
k_attention(const bf16_t* __restrict__ Q,   // [B,H,L,HD]
            const bf16_t* __restrict__ Kc,  // [B,KV,L,HD]
            const bf16_t* __restrict__ Vt,  // [B,KV,HD,L]
            bf16_t* __restrict__ O) {       // [B,L,H*HD]
    const int lane = threadIdx.x & 31;
    const int wid  = threadIdx.x >> 5;
    const int b = blockIdx.z, h = blockIdx.y;
    const int qbase = blockIdx.x * 64 + wid * 16;
    const int kvh = h / G_;
    const int r = lane & 15, khalf = lane >> 4;

    __shared__ __align__(16) bf16_t plds[4][16][34];

    ABFrag qfr[4];
    const bf16_t* qrow = Q + (((size_t)b * H_ + h) * L_ + qbase + r) * HD_;
    #pragma unroll
    for (int f = 0; f < 4; ++f)
        #pragma unroll
        for (int j = 0; j < 8; ++j) {
            const int ka = f * 32 + ((j < 4) ? (2 * j) : (16 + 2 * (j - 4))) + 8 * khalf;
            qfr[f].u[j] = *(const unsigned int*)(qrow + ka);
        }

    v8f oa[8];
    #pragma unroll
    for (int tt = 0; tt < 8; ++tt) oa[tt] = v8f_zero();
    float mrow[8], lrow[8];
    #pragma unroll
    for (int j = 0; j < 8; ++j) { mrow[j] = -1e30f; lrow[j] = 0.f; }

    const float scale = 0.08838834764831845f; // 1/sqrt(128)
    const bf16_t* kbp = Kc + ((size_t)b * KV_ + kvh) * (size_t)L_ * HD_;
    const bf16_t* vbp = Vt + ((size_t)b * KV_ + kvh) * (size_t)HD_ * L_;

    for (int kv0 = 0; kv0 < qbase + 16; kv0 += 32) {
        v8f s0 = v8f_zero(), s1 = v8f_zero();
        int kr0 = kv0 + r;      if (kr0 > L_ - 1) kr0 = L_ - 1;
        int kr1 = kv0 + 16 + r; if (kr1 > L_ - 1) kr1 = L_ - 1;
        const bf16_t* k0p = kbp + (size_t)kr0 * HD_;
        const bf16_t* k1p = kbp + (size_t)kr1 * HD_;
        #pragma unroll
        for (int f = 0; f < 4; ++f) {
            ABFrag b0, b1;
            #pragma unroll
            for (int j = 0; j < 8; ++j) {
                const int kk = f * 32 + ((j < 4) ? (2 * j) : (16 + 2 * (j - 4))) + 8 * khalf;
                b0.u[j] = *(const unsigned int*)(k0p + kk);
                b1.u[j] = *(const unsigned int*)(k1p + kk);
            }
            s0 = __builtin_amdgcn_wmma_f32_16x16x32_bf16(false, qfr[f].v, false, b0.v,
                                                         (short)0, s0, false, false);
            s1 = __builtin_amdgcn_wmma_f32_16x16x32_bf16(false, qfr[f].v, false, b1.v,
                                                         (short)0, s1, false, false);
        }
        #pragma unroll
        for (int j = 0; j < 8; ++j) {
            const int lm = j + 8 * khalf;
            const int qg = qbase + lm;
            const int n0 = kv0 + r, n1 = n0 + 16;
            float a0 = (n0 <= qg) ? s0[j] * scale : -1e30f;
            float a1 = (n1 <= qg) ? s1[j] * scale : -1e30f;
            float mx = fmaxf(a0, a1);
            #pragma unroll
            for (int off = 1; off < 16; off <<= 1) mx = fmaxf(mx, __shfl_xor(mx, off));
            const float mnew = fmaxf(mrow[j], mx);
            const float corr = __expf(mrow[j] - mnew);
            const float p0 = __expf(a0 - mnew);
            const float p1 = __expf(a1 - mnew);
            float rs = p0 + p1;
            #pragma unroll
            for (int off = 1; off < 16; off <<= 1) rs += __shfl_xor(rs, off);
            lrow[j] = lrow[j] * corr + rs;
            mrow[j] = mnew;
            #pragma unroll
            for (int tt = 0; tt < 8; ++tt) oa[tt][j] *= corr;
            plds[wid][lm][r]      = f2bf(p0);
            plds[wid][lm][16 + r] = f2bf(p1);
        }
        asm volatile("" ::: "memory"); // wave-local LDS is in-order; block compiler reorder
        ABFrag pf;
        #pragma unroll
        for (int j = 0; j < 8; ++j) {
            const int ka = ((j < 4) ? (2 * j) : (16 + 2 * (j - 4))) + 8 * khalf;
            pf.u[j] = *(const unsigned int*)&plds[wid][r][ka];
        }
        #pragma unroll
        for (int tt = 0; tt < 8; ++tt) {
            ABFrag vfr;
            #pragma unroll
            for (int j = 0; j < 8; ++j) {
                int kk = kv0 + ((j < 4) ? (2 * j) : (16 + 2 * (j - 4))) + 8 * khalf;
                if (kk > L_ - 2) kk = L_ - 2; // safe pad; masked P entries are 0
                vfr.u[j] = *(const unsigned int*)(vbp + (size_t)(tt * 16 + r) * L_ + kk);
            }
            oa[tt] = __builtin_amdgcn_wmma_f32_16x16x32_bf16(false, pf.v, false, vfr.v,
                                                             (short)0, oa[tt], false, false);
        }
        asm volatile("" ::: "memory");
    }
    #pragma unroll
    for (int j = 0; j < 8; ++j) {
        const int lm = j + 8 * khalf;
        const float inv = 1.f / lrow[j];
        const size_t orow = ((size_t)b * L_ + qbase + lm) * (size_t)(H_ * HD_) + (size_t)h * HD_;
        #pragma unroll
        for (int tt = 0; tt < 8; ++tt)
            O[orow + tt * 16 + r] = f2bf(oa[tt][j] * inv);
    }
}

// ---------------------------------------------------------------------------
// Router: logits = hm @ router_w^T, softmax(E=16), top-8, renormalize,
// scatter to dense [N, E]. One warp per token row.
// ---------------------------------------------------------------------------
__global__ void k_router(const bf16_t* __restrict__ hm, const bf16_t* __restrict__ rw,
                         float* __restrict__ wdense) {
    const int row  = blockIdx.x * 8 + (threadIdx.x >> 5);
    const int lane = threadIdx.x & 31;
    float acc[E_];
    #pragma unroll
    for (int e = 0; e < E_; ++e) acc[e] = 0.f;
    const bf16_t* hr = hm + (size_t)row * DM_;
    for (int k = lane; k < DM_; k += 32) {
        const float hv = bf2f(hr[k]);
        #pragma unroll
        for (int e = 0; e < E_; ++e) acc[e] += hv * bf2f(rw[(size_t)e * DM_ + k]);
    }
    #pragma unroll
    for (int e = 0; e < E_; ++e)
        #pragma unroll
        for (int off = 16; off > 0; off >>= 1) acc[e] += __shfl_xor(acc[e], off);
    if (lane == 0) {
        float mx = acc[0];
        for (int e = 1; e < E_; ++e) mx = fmaxf(mx, acc[e]);
        float p[E_]; float sum = 0.f;
        for (int e = 0; e < E_; ++e) { p[e] = __expf(acc[e] - mx); sum += p[e]; }
        for (int e = 0; e < E_; ++e) p[e] /= sum;
        float out[E_]; bool used[E_];
        for (int e = 0; e < E_; ++e) { out[e] = 0.f; used[e] = false; }
        float ssel = 0.f;
        for (int t = 0; t < TOPK_; ++t) {
            int bi = 0; float bv = -1.f;
            for (int e = 0; e < E_; ++e)
                if (!used[e] && p[e] > bv) { bv = p[e]; bi = e; }
            used[bi] = true; out[bi] = bv; ssel += bv;
        }
        for (int e = 0; e < E_; ++e) wdense[(size_t)row * E_ + e] = out[e] / ssel;
    }
}

// a = silu(g) * u  -> bf16
__global__ void k_silu_mul(const float* __restrict__ g, const float* __restrict__ u,
                           bf16_t* __restrict__ a, int n) {
    const int i = blockIdx.x * blockDim.x + threadIdx.x;
    if (i < n) {
        const float gv = g[i];
        a[i] = f2bf((gv / (1.f + __expf(-gv))) * u[i]);
    }
}

// ---------------------------------------------------------------------------
// Host-side orchestration
// ---------------------------------------------------------------------------
extern "C" void kernel_launch(void* const* d_in, const int* in_sizes, int n_in,
                              void* d_out, int out_size, void* d_ws, size_t ws_size,
                              hipStream_t stream) {
    (void)in_sizes; (void)n_in; (void)out_size; (void)ws_size;
    const float* x    = (const float*)d_in[0];
    const float* cosb = (const float*)d_in[1];
    const float* sinb = (const float*)d_in[2];
    const float* naw  = (const float*)d_in[3];
    const float* q_w  = (const float*)d_in[4];
    const float* k_w  = (const float*)d_in[5];
    const float* v_w  = (const float*)d_in[6];
    const float* qn_w = (const float*)d_in[7];
    const float* kn_w = (const float*)d_in[8];
    const float* o_w  = (const float*)d_in[9];
    const float* nmw  = (const float*)d_in[10];
    const float* r_w  = (const float*)d_in[11];
    const float* egw  = (const float*)d_in[12];
    const float* euw  = (const float*)d_in[13];
    const float* edw  = (const float*)d_in[14];
    float* out = (float*)d_out;

    size_t off = 0;
    auto alloc = [&](size_t bytes) -> char* {
        char* p = (char*)d_ws + off;
        off += (bytes + 255) & ~(size_t)255;
        return p;
    };
    bf16_t* hb  = (bf16_t*)alloc((size_t)N_ * DM_ * 2);
    bf16_t* qwb = (bf16_t*)alloc((size_t)H_ * HD_ * DM_ * 2);
    bf16_t* kwb = (bf16_t*)alloc((size_t)KV_ * HD_ * DM_ * 2);
    bf16_t* vwb = (bf16_t*)alloc((size_t)KV_ * HD_ * DM_ * 2);
    bf16_t* owb = (bf16_t*)alloc((size_t)DM_ * H_ * HD_ * 2);
    bf16_t* rwb = (bf16_t*)alloc((size_t)E_ * DM_ * 2);
    bf16_t* gwb = (bf16_t*)alloc((size_t)E_ * MI_ * DM_ * 2);
    bf16_t* uwb = (bf16_t*)alloc((size_t)E_ * MI_ * DM_ * 2);
    bf16_t* dwb = (bf16_t*)alloc((size_t)E_ * DM_ * MI_ * 2);
    float*  qf  = (float*)alloc((size_t)N_ * H_ * HD_ * 4);
    float*  kf  = (float*)alloc((size_t)N_ * KV_ * HD_ * 4);
    float*  vf  = (float*)alloc((size_t)N_ * KV_ * HD_ * 4);
    bf16_t* qb  = (bf16_t*)alloc((size_t)N_ * H_ * HD_ * 2);
    bf16_t* kb  = (bf16_t*)alloc((size_t)N_ * KV_ * HD_ * 2);
    bf16_t* vtb = (bf16_t*)alloc((size_t)N_ * KV_ * HD_ * 2);
    bf16_t* ob  = (bf16_t*)alloc((size_t)N_ * H_ * HD_ * 2);
    bf16_t* hmb = (bf16_t*)alloc((size_t)N_ * DM_ * 2);
    float*  wdn = (float*)alloc((size_t)N_ * E_ * 4);
    float*  gf  = (float*)alloc((size_t)N_ * MI_ * 4);
    float*  uf  = (float*)alloc((size_t)N_ * MI_ * 4);
    bf16_t* ab  = (bf16_t*)alloc((size_t)N_ * MI_ * 2);

    auto conv = [&](const float* src, bf16_t* dst, size_t n) {
        k_f32_to_bf16<<<(unsigned)((n + 255) / 256), 256, 0, stream>>>(src, dst, (int)n);
    };
    conv(q_w, qwb, (size_t)H_ * HD_ * DM_);
    conv(k_w, kwb, (size_t)KV_ * HD_ * DM_);
    conv(v_w, vwb, (size_t)KV_ * HD_ * DM_);
    conv(o_w, owb, (size_t)DM_ * H_ * HD_);
    conv(r_w, rwb, (size_t)E_ * DM_);
    conv(egw, gwb, (size_t)E_ * MI_ * DM_);
    conv(euw, uwb, (size_t)E_ * MI_ * DM_);
    conv(edw, dwb, (size_t)E_ * DM_ * MI_);

    // Blocked GEMM for large M (128x128 blocks, double-buffered async LDS staging)
    auto gemm_blk = [&](const bf16_t* A, const bf16_t* W, float* C, int Nr, int Mc, int K,
                        const float* residual, const float* rowscale, int rs_stride, int accum) {
        k_wmma_gemm_blk<<<dim3(Mc / 128, Nr / 128), 256, 0, stream>>>(
            A, W, C, Nr, Mc, K, residual, rowscale, rs_stride, accum);
    };
    // Simple GEMM for skinny M
    auto gemm_smp = [&](const bf16_t* A, const bf16_t* W, float* C, int Nr, int Mc, int K,
                        const float* residual, const float* rowscale, int rs_stride, int accum) {
        const int tiles = (Nr / 16) * (Mc / 16);
        k_wmma_gemm_nt<<<(tiles + 7) / 8, 256, 0, stream>>>(A, W, C, Nr, Mc, K,
                                                            residual, rowscale, rs_stride, accum);
    };

    // 1. attention-input RMSNorm -> bf16
    k_rmsnorm_bf16<<<N_, 256, 0, stream>>>(x, naw, hb);
    // 2. QKV projections (blocked)
    gemm_blk(hb, qwb, qf, N_, H_ * HD_, DM_, nullptr, nullptr, 0, 0);
    gemm_blk(hb, kwb, kf, N_, KV_ * HD_, DM_, nullptr, nullptr, 0, 0);
    gemm_blk(hb, vwb, vf, N_, KV_ * HD_, DM_, nullptr, nullptr, 0, 0);
    // 3. per-head norm + RoPE + relayout
    k_qk_norm_rope<<<dim3(N_, H_), HD_, 0, stream>>>(qf, qn_w, cosb, sinb, qb, H_);
    k_qk_norm_rope<<<dim3(N_, KV_), HD_, 0, stream>>>(kf, kn_w, cosb, sinb, kb, KV_);
    k_v_transpose<<<dim3(N_, KV_), HD_, 0, stream>>>(vf, vtb);
    // 4. flash attention (causal, GQA)
    k_attention<<<dim3(L_ / 64, H_, B_), 128, 0, stream>>>(qb, kb, vtb, ob);
    // 5. output projection + residual -> d_out (blocked)
    gemm_blk(ob, owb, out, N_, DM_, H_ * HD_, x, nullptr, 0, 0);
    // 6. MLP RMSNorm
    k_rmsnorm_bf16<<<N_, 256, 0, stream>>>(out, nmw, hmb);
    // 7. router (softmax + top-8 -> dense weights)
    k_router<<<N_ / 8, 256, 0, stream>>>(hmb, rwb, wdn);
    // 8. dense MoE: 16 experts, accumulate scaled down-proj into d_out
    for (int e = 0; e < E_; ++e) {
        const bf16_t* gw = gwb + (size_t)e * MI_ * DM_;
        const bf16_t* uw = uwb + (size_t)e * MI_ * DM_;
        const bf16_t* dw = dwb + (size_t)e * DM_ * MI_;
        gemm_smp(hmb, gw, gf, N_, MI_, DM_, nullptr, nullptr, 0, 0);
        gemm_smp(hmb, uw, uf, N_, MI_, DM_, nullptr, nullptr, 0, 0);
        k_silu_mul<<<(N_ * MI_ + 255) / 256, 256, 0, stream>>>(gf, uf, ab, N_ * MI_);
        gemm_blk(ab, dw, out, N_, DM_, MI_, nullptr, wdn + e, E_, 1);
    }
}
